// HoffmanSwarmV2_75625784148151
// MI455X (gfx1250) — matrix operand order repaired and checked by reading, not compiled, not run on hardware
//
#include <hip/hip_runtime.h>
#include <math.h>

// ---------------- problem constants ----------------
#define BB    4
#define NN    2048
#define DD    256
#define HH    2048
#define NCBC  256
#define KTOP  16
#define DP    64
#define QK_SCALE 8.0f   // sqrt(DP)
#define EPS   1e-8f
#define SLD   2056      // padded LDS row stride for the score slab

typedef __bf16 bf16_t;
typedef __attribute__((ext_vector_type(16))) __bf16 v16bf;
typedef __attribute__((ext_vector_type(8)))  float  v8f;

#if __has_builtin(__builtin_amdgcn_tensor_load_to_lds)
#define HAVE_TDM 1
typedef __attribute__((ext_vector_type(4))) unsigned int u32x4;
typedef __attribute__((ext_vector_type(8))) int i32x8;
typedef __attribute__((ext_vector_type(4))) int i32x4;
#endif

union FragU { v16bf v; uint4 q[2]; };

static __device__ __forceinline__ v8f wmma_bf16(v16bf a, v16bf b, v8f c) {
  // D = A(16x32 bf16) x B(32x16 bf16) + C(16x16 f32)
  return __builtin_amdgcn_wmma_f32_16x16x32_bf16(false, a, false, b, (short)0, c, false, false);
}
static __device__ __forceinline__ v8f zero8() {
  v8f z = {0.f,0.f,0.f,0.f,0.f,0.f,0.f,0.f};
  return z;
}

// A fragment (16x32, rows striped over lanes) from row-major bf16, ld elements.
static __device__ __forceinline__ v16bf load_a_bf16(const bf16_t* base, int ld,
                                                    int row0, int k0, int lane) {
  int half = lane >> 4, m = lane & 15;
  const bf16_t* p = base + (size_t)(row0 + m) * ld + k0 + half * 8;
  FragU f;
  f.q[0] = *reinterpret_cast<const uint4*>(p);       // k = k0+half*8 .. +7
  f.q[1] = *reinterpret_cast<const uint4*>(p + 16);  // k = k0+16+half*8 .. +7
  return f.v;
}

// Same A layout, source = LDS tile (row stride ld bf16) -> ds_load_b128 x2.
static __device__ __forceinline__ v16bf load_a_lds(const bf16_t* base, int ld,
                                                   int k0, int lane) {
  int half = lane >> 4, m = lane & 15;
  const bf16_t* p = base + m * ld + k0 + half * 8;
  FragU f;
  f.q[0] = *reinterpret_cast<const uint4*>(p);
  f.q[1] = *reinterpret_cast<const uint4*>(p + 16);
  return f.v;
}

// A fragment converted on the fly from row-major f32.
static __device__ __forceinline__ v16bf load_a_f32(const float* base, int ld,
                                                   int row0, int k0, int lane) {
  int half = lane >> 4, m = lane & 15;
  const float* p = base + (size_t)(row0 + m) * ld + k0 + half * 8;
  float4 x0 = *reinterpret_cast<const float4*>(p);
  float4 x1 = *reinterpret_cast<const float4*>(p + 4);
  float4 x2 = *reinterpret_cast<const float4*>(p + 16);
  float4 x3 = *reinterpret_cast<const float4*>(p + 20);
  v16bf r;
  r[0]=(bf16_t)x0.x;  r[1]=(bf16_t)x0.y;  r[2]=(bf16_t)x0.z;  r[3]=(bf16_t)x0.w;
  r[4]=(bf16_t)x1.x;  r[5]=(bf16_t)x1.y;  r[6]=(bf16_t)x1.z;  r[7]=(bf16_t)x1.w;
  r[8]=(bf16_t)x2.x;  r[9]=(bf16_t)x2.y;  r[10]=(bf16_t)x2.z; r[11]=(bf16_t)x2.w;
  r[12]=(bf16_t)x3.x; r[13]=(bf16_t)x3.y; r[14]=(bf16_t)x3.z; r[15]=(bf16_t)x3.w;
  return r;
}

// B fragment (32x16 K x N) from N-major bf16 (B^T row-major (N,K)).
static __device__ __forceinline__ v16bf load_bt_bf16(const bf16_t* bt, int ld,
                                                     int n0, int k0, int lane) {
  int half = lane >> 4, n = lane & 15;
  const bf16_t* p = bt + (size_t)(n0 + n) * ld + k0 + half * 8;
  FragU f;
  f.q[0] = *reinterpret_cast<const uint4*>(p);
  f.q[1] = *reinterpret_cast<const uint4*>(p + 16);
  return f.v;
}

// B fragment gathered from row-major f32 where K indexes rows (stride ld).
static __device__ __forceinline__ v16bf load_b_rows_f32(const float* src, int ld,
                                                        int k0, int n0, int lane) {
  int half = lane >> 4, n = lane & 15;
  const float* p = src + (size_t)(k0 + half * 8) * ld + n0 + n;
  v16bf r;
  #pragma unroll
  for (int e = 0; e < 16; e++) r[e] = (bf16_t)p[(size_t)(e + (e & 8)) * ld];
  return r;
}

// ---------------- kernels ----------------

// Weight prep: src (K,Nc) f32 row-major -> dst (Nc,K) bf16 row-major.
__global__ void k_transpose_bf16(const float* __restrict__ src, bf16_t* __restrict__ dst,
                                 int K, int Nc) {
  int i = blockIdx.x * blockDim.x + threadIdx.x;
  if (i < K * Nc) {
    int n = i / K, k = i - n * K;
    dst[(size_t)n * K + k] = (bf16_t)src[(size_t)k * Nc + n];
  }
}

// Q/K/proj projections: (B*N,256) x (256,64). blockIdx.y selects matrix.
// Two independent WMMA accumulator chains hide the D->C RAW latency.
__global__ void k_qkp(const float* __restrict__ actions,
                      const bf16_t* __restrict__ WqT, const bf16_t* __restrict__ WkT,
                      const bf16_t* __restrict__ WcT,
                      const float* __restrict__ bq, const float* __restrict__ bk,
                      const float* __restrict__ bc,
                      bf16_t* __restrict__ Qbf, bf16_t* __restrict__ Kbf,
                      float* __restrict__ projf) {
  int row0 = blockIdx.x * 16;
  int which = blockIdx.y;
  int wave = threadIdx.x >> 5, lane = threadIdx.x & 31;
  const bf16_t* WT = (which == 0) ? WqT : (which == 1) ? WkT : WcT;
  const float* bias = (which == 0) ? bq : (which == 1) ? bk : bc;
  int c0 = wave * 16;
  v8f acc0 = zero8(), acc1 = zero8();
  #pragma unroll
  for (int s = 0; s < 8; s += 2) {
    v16bf a0 = load_a_f32(actions, DD, row0, 32 * s, lane);
    v16bf b0 = load_bt_bf16(WT, DD, c0, 32 * s, lane);
    v16bf a1 = load_a_f32(actions, DD, row0, 32 * (s + 1), lane);
    v16bf b1 = load_bt_bf16(WT, DD, c0, 32 * (s + 1), lane);
    acc0 = wmma_bf16(a0, b0, acc0);
    acc1 = wmma_bf16(a1, b1, acc1);
  }
  v8f acc = acc0 + acc1;
  int cc = lane & 15, half = lane >> 4;
  float bv = bias[c0 + cc];
  #pragma unroll
  for (int v = 0; v < 8; v++) {
    int r = v + 8 * half;
    float val = acc[v] + bv;
    size_t idx = (size_t)(row0 + r) * DP + c0 + cc;
    if (which == 0)      Qbf[idx] = (bf16_t)val;
    else if (which == 1) Kbf[idx] = (bf16_t)val;
    else                 projf[idx] = val;
  }
}

// Row-normalize proj -> pn (bf16). One wave per row of 64.
__global__ void k_pnorm(const float* __restrict__ projf, bf16_t* __restrict__ pnbf) {
  int row = blockIdx.x * 4 + (threadIdx.x >> 5);
  int lane = threadIdx.x & 31;
  const float* p = projf + (size_t)row * DP;
  float a = p[lane], b = p[lane + 32];
  float ss = a * a + b * b;
  #pragma unroll
  for (int off = 16; off; off >>= 1) ss += __shfl_xor(ss, off, 32);
  float inv = 1.f / fmaxf(sqrtf(ss), 1e-12f);
  pnbf[(size_t)row * DP + lane]      = (bf16_t)(a * inv);
  pnbf[(size_t)row * DP + lane + 32] = (bf16_t)(b * inv);
}

// Dynamic-graph: per 16-row tile, WMMA full 16x2048 score slab into LDS
// (CDNA5 320KB/WGP), exact top-16 + softmax + sparse gather of incoming.
__global__ void k_graph(const bf16_t* __restrict__ Qbf, const bf16_t* __restrict__ Kbf,
                        const float* __restrict__ states, const float* __restrict__ log_temp,
                        float* __restrict__ incoming) {
  extern __shared__ float sc[];  // [16][SLD] f32 = ~128.5 KB
  int tile = blockIdx.x;
  int b = tile / (NN / 16);
  int n0 = (tile % (NN / 16)) * 16;
  int wave = threadIdx.x >> 5, lane = threadIdx.x & 31;
  int cc = lane & 15, half = lane >> 4;
  const bf16_t* Qb = Qbf + (size_t)b * NN * DP;
  const bf16_t* Kb = Kbf + (size_t)b * NN * DP;
  v16bf a0 = load_a_bf16(Qb, DP, n0, 0, lane);
  v16bf a1 = load_a_bf16(Qb, DP, n0, 32, lane);
  for (int t = 0; t < 32; t++) {
    int m0 = (wave * 32 + t) * 16;
    v16bf b0 = load_bt_bf16(Kb, DP, m0, 0, lane);
    v16bf b1 = load_bt_bf16(Kb, DP, m0, 32, lane);
    v8f d = zero8();
    d = wmma_bf16(a0, b0, d);
    d = wmma_bf16(a1, b1, d);
    #pragma unroll
    for (int v = 0; v < 8; v++) sc[(v + 8 * half) * SLD + m0 + cc] = d[v];
  }
  __syncthreads();
  float temp = fminf(fmaxf(expf(log_temp[0]), 0.1f), 10.f);
  float invs = 1.f / (QK_SCALE * temp);
  for (int rr = 0; rr < 4; rr++) {
    int r = wave * 4 + rr;                 // each wave owns 4 rows
    float* row = sc + r * SLD;
    float vals[KTOP]; int idxs[KTOP];
    for (int t = 0; t < KTOP; t++) {
      float best = -3.0e38f; int bi = 0;
      for (int j = lane; j < NN; j += 32) {
        float v = row[j];
        if (v > best) { best = v; bi = j; }
      }
      #pragma unroll
      for (int off = 16; off; off >>= 1) {
        float ov = __shfl_xor(best, off, 32);
        int   oi = __shfl_xor(bi, off, 32);
        if (ov > best || (ov == best && oi < bi)) { best = ov; bi = oi; }
      }
      vals[t] = best; idxs[t] = bi;
      if (lane == 0) row[bi] = -3.0e38f;   // LDS in-order within wave
    }
    float wsum = 0.f, wgt[KTOP];
    #pragma unroll
    for (int t = 0; t < KTOP; t++) { wgt[t] = expf((vals[t] - vals[0]) * invs); wsum += wgt[t]; }
    float winv = 1.f / wsum;
    const float* Sb = states + (size_t)b * NN * DD;
    float acc[8] = {0,0,0,0,0,0,0,0};
    for (int t = 0; t < KTOP; t++) {
      const float* sr = Sb + (size_t)idxs[t] * DD + lane * 8;
      float ww = wgt[t] * winv;
      float4 x0 = *reinterpret_cast<const float4*>(sr);
      float4 x1 = *reinterpret_cast<const float4*>(sr + 4);
      acc[0] += ww * x0.x; acc[1] += ww * x0.y; acc[2] += ww * x0.z; acc[3] += ww * x0.w;
      acc[4] += ww * x1.x; acc[5] += ww * x1.y; acc[6] += ww * x1.z; acc[7] += ww * x1.w;
    }
    float* op = incoming + ((size_t)b * NN + n0 + r) * DD + lane * 8;
    #pragma unroll
    for (int j = 0; j < 8; j++) op[j] = acc[j];
  }
}

// Coalition: streaming sim -> sigmoid -> cw (staged via per-wave LDS tile to
// re-layout into an A fragment) -> WMMA accumulate cw@incoming; then mixed.
__global__ void k_coalition(const bf16_t* __restrict__ pnbf, const float* __restrict__ incoming,
                            const float* __restrict__ temp_coal, float* __restrict__ mixed) {
  __shared__ __align__(16) bf16_t cwlds[4][16][40];
  int tile = blockIdx.x;
  int b = tile / (NN / 16);
  int n0 = (tile % (NN / 16)) * 16;
  int wave = threadIdx.x >> 5, lane = threadIdx.x & 31;
  int cc = lane & 15, half = lane >> 4;
  const bf16_t* Pb = pnbf + (size_t)b * NN * DP;
  const float* Ib = incoming + (size_t)b * NN * DD;
  float tc = temp_coal[0];
  v16bf a0 = load_a_bf16(Pb, DP, n0, 0, lane);
  v16bf a1 = load_a_bf16(Pb, DP, n0, 32, lane);
  int d0w = wave * 64;                       // each wave owns 64 cols of D
  v8f comb[4] = {zero8(), zero8(), zero8(), zero8()};
  float rs[8] = {0,0,0,0,0,0,0,0};
  bf16_t* myc = &cwlds[wave][0][0];
  for (int scnk = 0; scnk < NN / 32; scnk++) {
    int mbase = scnk * 32;
    #pragma unroll
    for (int hseg = 0; hseg < 2; hseg++) {
      int m0 = mbase + hseg * 16;
      v16bf b0 = load_bt_bf16(Pb, DP, m0, 0, lane);
      v16bf b1 = load_bt_bf16(Pb, DP, m0, 32, lane);
      v8f d = zero8();
      d = wmma_bf16(a0, b0, d);
      d = wmma_bf16(a1, b1, d);
      #pragma unroll
      for (int v = 0; v < 8; v++) {
        float cw = 1.f / (1.f + expf(-(d[v] - 0.7f) * tc));
        rs[v] += cw;
        myc[(v + 8 * half) * 40 + hseg * 16 + cc] = (bf16_t)cw;
      }
    }
    v16bf ca = load_a_lds(myc, 40, 0, lane);  // cw as 16x32 A fragment
    #pragma unroll
    for (int ct = 0; ct < 4; ct++) {
      v16bf bi = load_b_rows_f32(Ib, DD, mbase, d0w + ct * 16, lane);
      comb[ct] = wmma_bf16(ca, bi, comb[ct]);
    }
  }
  #pragma unroll
  for (int off = 1; off < 16; off <<= 1) {
    #pragma unroll
    for (int v = 0; v < 8; v++) rs[v] += __shfl_xor(rs[v], off, 32);
  }
  #pragma unroll
  for (int ct = 0; ct < 4; ct++) {
    #pragma unroll
    for (int v = 0; v < 8; v++) {
      int r = v + 8 * half;
      int dcol = d0w + ct * 16 + cc;
      size_t li = (size_t)(n0 + r) * DD + dcol;
      float m = 0.8f * Ib[li] + 0.2f * (comb[ct][v] / (rs[v] + EPS));
      mixed[(size_t)b * NN * DD + li] = m;
    }
  }
}

// HDC: mixed@W_to_hdc -> sign (STE fwd) -> bind/write/cosine accumulators.
__global__ void k_hdc(const float* __restrict__ mixed, const float* __restrict__ episodic,
                      const bf16_t* __restrict__ WtoT, const float* __restrict__ b_to,
                      const float* __restrict__ keys, const float* __restrict__ pos_codes,
                      const int* __restrict__ step, bf16_t* __restrict__ newmem,
                      float* __restrict__ strength) {
  __shared__ float s_npq[16], s_nn[16], s_pp[16];
  int tile = blockIdx.x;
  int b = tile / (NN / 16);
  int n0 = (tile % (NN / 16)) * 16;
  int wave = threadIdx.x >> 5, lane = threadIdx.x & 31;
  int cc = lane & 15, half = lane >> 4;
  if (threadIdx.x < 16) { s_npq[threadIdx.x] = 0.f; s_nn[threadIdx.x] = 0.f; s_pp[threadIdx.x] = 0.f; }
  __syncthreads();
  const float* Mb = mixed + (size_t)b * NN * DD;
  v16bf a[8];
  #pragma unroll
  for (int s = 0; s < 8; s++) a[s] = load_a_f32(Mb, DD, n0, 32 * s, lane);
  const float* pc = pos_codes + (size_t)(step[0] % NCBC) * HH;
  float d_npq[8] = {0,0,0,0,0,0,0,0};
  float d_nn[8]  = {0,0,0,0,0,0,0,0};
  float d_pp[8]  = {0,0,0,0,0,0,0,0};
  for (int t = 0; t < 32; t++) {
    int h0 = wave * 512 + t * 16;           // each wave owns 512 of H
    v8f acc0 = zero8(), acc1 = zero8();
    #pragma unroll
    for (int s = 0; s < 8; s += 2) {        // two independent WMMA chains
      acc0 = wmma_bf16(a[s],     load_bt_bf16(WtoT, DD, h0, 32 * s,      lane), acc0);
      acc1 = wmma_bf16(a[s + 1], load_bt_bf16(WtoT, DD, h0, 32 * (s + 1), lane), acc1);
    }
    v8f acc = acc0 + acc1;
    int h = h0 + cc;
    float bt = b_to[h];
    float pcv = pc[h];
    #pragma unroll
    for (int v = 0; v < 8; v++) {
      int n = n0 + v + 8 * half;
      float pre = acc[v] + bt;
      float q = (pre >= 0.f) ? 1.f : -1.f;  // sign with sign(0)->+1
      float key = keys[(size_t)n * HH + h];
      float mem = episodic[((size_t)b * NN + n) * HH + h];
      float nm = 0.95f * mem + 0.05f * q * pcv;
      float pq = q * key;
      d_npq[v] += nm * pq;
      d_nn[v]  += nm * nm;
      d_pp[v]  += pq * pq;
      newmem[((size_t)b * NN + n) * HH + h] = (bf16_t)nm;
    }
  }
  #pragma unroll
  for (int off = 1; off < 16; off <<= 1) {
    #pragma unroll
    for (int v = 0; v < 8; v++) {
      d_npq[v] += __shfl_xor(d_npq[v], off, 32);
      d_nn[v]  += __shfl_xor(d_nn[v],  off, 32);
      d_pp[v]  += __shfl_xor(d_pp[v],  off, 32);
    }
  }
  if (cc == 0) {
    #pragma unroll
    for (int v = 0; v < 8; v++) {
      int r = v + 8 * half;
      atomicAdd(&s_npq[r], d_npq[v]);       // ds_add_f32
      atomicAdd(&s_nn[r],  d_nn[v]);
      atomicAdd(&s_pp[r],  d_pp[v]);
    }
  }
  __syncthreads();
  if (threadIdx.x < 16) {
    int r = threadIdx.x;
    float na = fmaxf(sqrtf(s_nn[r]), EPS);
    float nb = fmaxf(sqrtf(s_pp[r]), EPS);
    float cosv = s_npq[r] / (na * nb);
    strength[(size_t)b * NN + n0 + r] = 1.f / (1.f + expf(-cosv));
  }
}

// Recall: new_mem(bf16) @ W_from_hdc^T(bf16) over K=2048.
// The 16x2048 bf16 A-slab (64KB) is DMA'd into LDS with one Tensor-Data-Mover
// tensor_load_to_lds per workgroup (TENSORcnt / s_wait_tensorcnt), then the
// K-loop reads A fragments via ds_load_b128 while B streams from L2.
__global__ void k_recall(const bf16_t* __restrict__ newmem, const bf16_t* __restrict__ WfromT,
                         const float* __restrict__ b_from, const float* __restrict__ strength,
                         const float* __restrict__ mixed, float* __restrict__ out) {
  __shared__ __align__(16) bf16_t atile[16 * HH];   // 64 KB of the 320 KB WGP LDS
  int tile = blockIdx.x;
  int b = tile / (NN / 16);
  int n0 = (tile % (NN / 16)) * 16;
  int wave = threadIdx.x >> 5, lane = threadIdx.x & 31;
  int cc = lane & 15, half = lane >> 4;
  const bf16_t* Ab = newmem + (size_t)b * NN * HH;

#ifdef HAVE_TDM
  if (threadIdx.x < 32) {  // wave 0 issues the TDM DMA (EXEC ignored by TDM)
    unsigned lds_addr = (unsigned)(unsigned long long)(size_t)&atile[0];
    unsigned long long ga = (unsigned long long)(size_t)(Ab + (size_t)n0 * HH);
    // D# group 0: count=1 | lds_addr | global_addr | type=2 ("image")
    u32x4 g0 = { 1u,
                 lds_addr,
                 (unsigned)(ga & 0xffffffffull),
                 (unsigned)((ga >> 32) & 0x01ffffffull) | (2u << 30) };
    // D# group 1: data_size=2B; tensor = 2048 x 16; tile = 2048 x 16; dim0 stride = 2048
    i32x8 g1 = { (int)(1u << 16),                 // workgroup_mask=0, data_size=1 (2B)
                 (int)((HH & 0xffff) << 16),      // tensor_dim0[15:0]
                 (int)(16 << 16),                 // tensor_dim1[15:0]
                 (int)((HH & 0xffff) << 16),      // tile_dim0
                 16,                              // tile_dim1=16, tile_dim2=0
                 HH,                              // tensor_dim0_stride[31:0]
                 0, 0 };
    i32x4 gz = {0, 0, 0, 0};                      // groups 2/3 unused (2D tile)
#if defined(__clang_major__) && __clang_major__ >= 23
    i32x8 gz8 = {0, 0, 0, 0, 0, 0, 0, 0};
    __builtin_amdgcn_tensor_load_to_lds(g0, g1, gz, gz, gz8, 0);
#else
    __builtin_amdgcn_tensor_load_to_lds(g0, g1, gz, gz, 0);
#endif
#if __has_builtin(__builtin_amdgcn_s_wait_tensorcnt)
    __builtin_amdgcn_s_wait_tensorcnt(0);
#endif
  }
  __syncthreads();
#else
  // Fallback: cooperative vectorized copy (slab is contiguous in memory).
  {
    const uint4* src = reinterpret_cast<const uint4*>(Ab + (size_t)n0 * HH);
    uint4* dst = reinterpret_cast<uint4*>(&atile[0]);
    for (int i = threadIdx.x; i < (16 * HH * 2) / 16; i += blockDim.x) dst[i] = src[i];
  }
  __syncthreads();
#endif

  int d0w = wave * 64;
  v8f acc[4] = {zero8(), zero8(), zero8(), zero8()};
  for (int ks = 0; ks < HH / 32; ks++) {
    v16bf a = load_a_lds(atile, HH, ks * 32, lane);
    if (ks + 8 < HH / 32)
      __builtin_prefetch(WfromT + (size_t)(d0w + (lane & 15)) * HH + (ks + 8) * 32, 0, 0);
    #pragma unroll
    for (int ct = 0; ct < 4; ct++)
      acc[ct] = wmma_bf16(a, load_bt_bf16(WfromT, HH, d0w + ct * 16, ks * 32, lane), acc[ct]);
  }
  #pragma unroll
  for (int ct = 0; ct < 4; ct++) {
    #pragma unroll
    for (int v = 0; v < 8; v++) {
      int n = n0 + v + 8 * half;
      int dcol = d0w + ct * 16 + cc;
      size_t gi = ((size_t)b * NN + n) * DD + dcol;
      float s = strength[(size_t)b * NN + n];
      out[gi] = mixed[gi] + (acc[ct][v] + b_from[dcol]) * s;
    }
  }
}

// ---------------- launch ----------------
extern "C" void kernel_launch(void* const* d_in, const int* in_sizes, int n_in,
                              void* d_out, int out_size, void* d_ws, size_t ws_size,
                              hipStream_t stream) {
  const float* states    = (const float*)d_in[0];
  const float* actions   = (const float*)d_in[1];
  const float* episodic  = (const float*)d_in[2];
  const float* Wq        = (const float*)d_in[3];
  const float* bq        = (const float*)d_in[4];
  const float* Wk        = (const float*)d_in[5];
  const float* bk        = (const float*)d_in[6];
  const float* log_temp  = (const float*)d_in[7];
  const float* Wc        = (const float*)d_in[8];
  const float* bc        = (const float*)d_in[9];
  const float* temp_coal = (const float*)d_in[10];
  const float* Wto       = (const float*)d_in[11];
  const float* b_to      = (const float*)d_in[12];
  const float* Wfrom     = (const float*)d_in[13];
  const float* b_from    = (const float*)d_in[14];
  const float* keys      = (const float*)d_in[15];
  const float* pos       = (const float*)d_in[16];
  const int*   step      = (const int*)d_in[17];
  float* out = (float*)d_out;

  char* w = (char*)d_ws;
  const size_t MB = 1ull << 20;
  bf16_t* Qbf      = (bf16_t*)(w + 0);                 // 1 MB
  bf16_t* Kbf      = (bf16_t*)(w + 1 * MB);            // 1 MB
  bf16_t* Pnbf     = (bf16_t*)(w + 2 * MB);            // 1 MB
  float*  projf    = (float*)(w + 3 * MB);             // 2 MB
  bf16_t* WqT      = (bf16_t*)(w + 5 * MB);            // 32 KB
  bf16_t* WkT      = (bf16_t*)(w + 5 * MB + 32768);    // 32 KB
  bf16_t* WcT      = (bf16_t*)(w + 5 * MB + 65536);    // 32 KB
  bf16_t* WtoT     = (bf16_t*)(w + 5 * MB + 98304);    // 1 MB
  bf16_t* WfromT   = (bf16_t*)(w + 6 * MB + 98304);    // 1 MB
  float*  incoming = (float*)(w + 7 * MB + 98304);     // 8 MB
  float*  mixed    = (float*)(w + 15 * MB + 98304);    // 8 MB
  bf16_t* newmem   = (bf16_t*)(w + 23 * MB + 98304);   // 32 MB
  float*  strength = (float*)(w + 55 * MB + 98304);    // 32 KB

  dim3 blk(128);  // 4 waves (wave32)
  k_transpose_bf16<<<(DD * DP + 255) / 256, 256, 0, stream>>>(Wq, WqT, DD, DP);
  k_transpose_bf16<<<(DD * DP + 255) / 256, 256, 0, stream>>>(Wk, WkT, DD, DP);
  k_transpose_bf16<<<(DD * DP + 255) / 256, 256, 0, stream>>>(Wc, WcT, DD, DP);
  k_transpose_bf16<<<(DD * HH + 255) / 256, 256, 0, stream>>>(Wto, WtoT, DD, HH);
  k_transpose_bf16<<<(HH * DD + 255) / 256, 256, 0, stream>>>(Wfrom, WfromT, HH, DD);

  int ntiles = (BB * NN) / 16;  // 512
  k_qkp<<<dim3(ntiles, 3), blk, 0, stream>>>(actions, WqT, WkT, WcT, bq, bk, bc,
                                             Qbf, Kbf, projf);
  k_pnorm<<<(BB * NN) / 4, blk, 0, stream>>>(projf, Pnbf);

  int smem = 16 * SLD * 4;  // 131,584 B — legal on CDNA5 (320 KB LDS/WGP)
  hipFuncSetAttribute((const void*)k_graph, hipFuncAttributeMaxDynamicSharedMemorySize, smem);
  k_graph<<<ntiles, blk, smem, stream>>>(Qbf, Kbf, states, log_temp, incoming);

  k_coalition<<<ntiles, blk, 0, stream>>>(Pnbf, incoming, temp_coal, mixed);
  k_hdc<<<ntiles, blk, 0, stream>>>(mixed, episodic, WtoT, b_to, keys, pos, step,
                                    newmem, strength);
  k_recall<<<ntiles, blk, 0, stream>>>(newmem, WfromT, b_from, strength, mixed, out);

  (void)in_sizes; (void)n_in; (void)out_size; (void)ws_size;
}
